// BaselineAttention_19670950216027
// MI455X (gfx1250) — compile-verified, matrix operand order checked by reading, not run on hardware
//
#include <hip/hip_runtime.h>
#include <hip/hip_bf16.h>

typedef __bf16 bf16;
typedef __attribute__((ext_vector_type(16))) __bf16 v16bf;
typedef __attribute__((ext_vector_type(8))) float v8f;
typedef uint32_t u32x4 __attribute__((ext_vector_type(4)));
typedef uint32_t u32x8 __attribute__((ext_vector_type(8)));

#define DMODEL 1024
#define NHEAD 16
#define SEQ 2048
#define BATCH 2
#define MROWS (BATCH * SEQ)  // 4096
#define MB ((size_t)1 << 20)

__device__ __forceinline__ v8f wmma_bf16(v16bf a, v16bf b, v8f c) {
  return __builtin_amdgcn_wmma_f32_16x16x32_bf16(
      false, a, false, b, (short)0, c, false, false);
}

// 16x32 bf16 fragment from LDS [16+][ld]: lane L -> row L&15, K-chunk (L>>4)*16.
__device__ __forceinline__ v16bf frag_ld(const bf16* p0, int ld, int lane) {
  const bf16* p = p0 + (lane & 15) * ld + ((lane >> 4) << 4);
  return *(const v16bf*)p;
}

// --- CDNA5 async copy (per-lane, ASYNCcnt) ---------------------------------
__device__ __forceinline__ void async_copy_b128(void* lds_ptr, const void* gptr) {
  uint32_t loff = (uint32_t)(uintptr_t)lds_ptr;
  asm volatile("global_load_async_to_lds_b128 %0, %1, off"
               :: "v"(loff), "v"(gptr)
               : "memory");
}
__device__ __forceinline__ void wait_async() {
  asm volatile("s_wait_asynccnt 0x0" ::: "memory");
}
__device__ __forceinline__ void lds_fence() {  // stop compiler reordering only
  asm volatile("" ::: "memory");
}

// --- CDNA5 Tensor Data Mover: 2D tile global -> LDS (TENSORcnt) ------------
// D# per ISA 8.3/8.4: group0 {count=1, lds_addr, global_addr, type=2},
// group1 {wg_mask=0, data_size=1 (2B), tensor dims, tile dims, dim0 stride}.
__device__ __forceinline__ void tdm_load_2d(const void* gptr, void* lds_ptr,
                                            uint32_t tensor_d0,
                                            uint32_t tensor_d1,
                                            uint32_t tile_d0, uint32_t tile_d1,
                                            uint32_t row_stride_elems) {
  const uint64_t ga = (uint64_t)(uintptr_t)gptr;
  const uint32_t la = (uint32_t)(uintptr_t)lds_ptr;
  u32x4 g0;
  g0[0] = 1u;                                  // count=1 (valid), user mode
  g0[1] = la;                                  // lds_addr (bytes)
  g0[2] = (uint32_t)ga;                        // global_addr[31:0]
  g0[3] = (uint32_t)(ga >> 32) | (2u << 30);   // global_addr[56:32] | type=2
  u32x8 g1;
  g1[0] = (1u << 16);                          // wg_mask=0, data_size=1 (2B)
  g1[1] = (tensor_d0 & 0xffffu) << 16;         // tensor_dim0[15:0]
  g1[2] = (tensor_d0 >> 16) | ((tensor_d1 & 0xffffu) << 16);
  g1[3] = (tensor_d1 >> 16) | (tile_d0 << 16); // tile_dim0
  g1[4] = tile_d1;                             // tile_dim1 (tile_dim2=0)
  g1[5] = row_stride_elems;                    // tensor_dim0_stride[31:0]
  g1[6] = 0;
  g1[7] = 0;
  asm volatile("tensor_load_to_lds %0, %1"
               :: "s"(g0), "s"(g1)
               : "memory");
}
__device__ __forceinline__ void wait_tensor() {
  __builtin_amdgcn_s_wait_tensorcnt(0);
}

// ---------------------------------------------------------------------------
// Prep: f32 -> bf16 elementwise, and W[K,N] f32 -> Wt[N,K] bf16 (transposed).
// ---------------------------------------------------------------------------
__global__ __launch_bounds__(256) void cvt_f32_bf16(
    const float* __restrict__ in, bf16* __restrict__ out, int n) {
  const int i = blockIdx.x * 256 + threadIdx.x;
  if (i < n) out[i] = (bf16)in[i];
}

__global__ __launch_bounds__(256) void transpose_f32_bf16(
    const float* __restrict__ W, bf16* __restrict__ Wt, int K, int N) {
  const int i = blockIdx.x * 256 + threadIdx.x;  // i = n*K + kk
  if (i < K * N) {
    const int n = i / K, kk = i - n * K;
    Wt[i] = (bf16)W[(size_t)kk * N + n];
  }
}

// ---------------------------------------------------------------------------
// GEMM: Y = A[M,K](bf16) @ Wt[N,K]^T(bf16) + bias.
// Tiles fetched by the Tensor Data Mover (one 128x32 descriptor per tile),
// double-buffered in LDS. Block tile 128x128, 256 thr (8 waves),
// wave tile 32x64 (2x4 WMMA).
// VT=true stores V transposed: vT[(b*1024 + col)*2048 + s], row = b*2048+s.
// ---------------------------------------------------------------------------
template <typename OT, bool VT>
__global__ __launch_bounds__(256) void gemm_tdm(
    const bf16* __restrict__ A, const bf16* __restrict__ Wt,
    const float* __restrict__ bias, OT* __restrict__ Y, int N, int K) {
  __shared__ bf16 lds_a[2][128 * 32];
  __shared__ bf16 lds_b[2][128 * 32];

  const int t = threadIdx.x;
  const int lane = t & 31;
  const int wave = t >> 5;
  const int mBase = blockIdx.y * 128;
  const int nBase = blockIdx.x * 128;
  const int wr = wave >> 1, wc = wave & 1;

  auto issue = [&](int kt, int bi) {
    if (wave == 0) {  // TDM is wave-level; EXEC ignored
      tdm_load_2d(A + (size_t)mBase * K + kt, &lds_a[bi][0],
                  (uint32_t)K, (uint32_t)MROWS, 32u, 128u, (uint32_t)K);
      tdm_load_2d(Wt + (size_t)nBase * K + kt, &lds_b[bi][0],
                  (uint32_t)K, (uint32_t)N, 32u, 128u, (uint32_t)K);
    }
  };

  v8f zero = {};
  v8f acc[2][4];
#pragma unroll
  for (int mi = 0; mi < 2; ++mi)
#pragma unroll
    for (int nj = 0; nj < 4; ++nj) acc[mi][nj] = zero;

  issue(0, 0);
  if (wave == 0) wait_tensor();
  __syncthreads();

  const int nk = K / 32;
  for (int kt = 0; kt < nk; ++kt) {
    const int bi = kt & 1;
    if (kt + 1 < nk) issue((kt + 1) * 32, bi ^ 1);

    v16bf af[2], bfr[4];
#pragma unroll
    for (int mi = 0; mi < 2; ++mi)
      af[mi] = frag_ld(&lds_a[bi][(wr * 32 + mi * 16) * 32], 32, lane);
#pragma unroll
    for (int nj = 0; nj < 4; ++nj)
      bfr[nj] = frag_ld(&lds_b[bi][(wc * 64 + nj * 16) * 32], 32, lane);
#pragma unroll
    for (int mi = 0; mi < 2; ++mi)
#pragma unroll
      for (int nj = 0; nj < 4; ++nj)
        acc[mi][nj] = wmma_bf16(af[mi], bfr[nj], acc[mi][nj]);

    if (wave == 0) wait_tensor();
    __syncthreads();
  }

  // C layout: VGPR r, lanes 0-15 -> M=r, lanes 16-31 -> M=8+r; N = lane&15.
  const int mhalf = (lane >> 4) << 3;
  const int ncol = lane & 15;
#pragma unroll
  for (int mi = 0; mi < 2; ++mi)
#pragma unroll
    for (int nj = 0; nj < 4; ++nj) {
      const int col = nBase + wc * 64 + nj * 16 + ncol;
      const float bv = bias[col];
#pragma unroll
      for (int r = 0; r < 8; ++r) {
        const int row = mBase + wr * 32 + mi * 16 + mhalf + r;
        const float val = acc[mi][nj][r] + bv;
        if constexpr (VT) {
          const size_t bi2 = (size_t)(row >> 11);
          const size_t s = (size_t)(row & 2047);
          Y[((bi2 * 1024 + col) << 11) + s] = (OT)val;
        } else {
          Y[(size_t)row * N + col] = (OT)val;
        }
      }
    }
}

// ---------------------------------------------------------------------------
// Flash attention. Grid (S/64, H, B), 128 thr = 4 waves; wave owns 16 q-rows.
// q,k bf16 [B*S,512]; vT bf16 [B*16*64, 2048]; out f32 [B*S,1024].
// K and V tiles via per-lane async-to-LDS, double-buffered.
// ---------------------------------------------------------------------------
__global__ __launch_bounds__(128) void attn_wmma(
    const bf16* __restrict__ q, const bf16* __restrict__ k,
    const bf16* __restrict__ vT, float* __restrict__ out) {
  __shared__ bf16 lds_k[2][64 * 32];   // [key][dq]
  __shared__ bf16 lds_v[2][64 * 64];   // [d][kv]
  __shared__ bf16 lds_p[4][16 * 64];   // per-wave P [m][kv]

  const int t = threadIdx.x, lane = t & 31, wave = t >> 5;
  const int h = blockIdx.y, b = blockIdx.z;
  const int q0 = blockIdx.x * 64;
  const int mhalf = (lane >> 4) << 3;
  const int ncol = lane & 15;

  const size_t qrow = (size_t)(b * SEQ + q0 + wave * 16 + (lane & 15));
  const v16bf qf = *(const v16bf*)(q + qrow * 512 + h * 32 + ((lane >> 4) << 4));

  const int kc_row = t >> 1, kc_c = (t & 1) << 4;  // 64x32 tile
  const int vc_d = t >> 1, vc_c = (t & 1) << 5;    // 64x64 tile

  auto issue = [&](int kv0, int bi) {
    const bf16* gk =
        k + (size_t)(b * SEQ + kv0 + kc_row) * 512 + h * 32 + kc_c;
    async_copy_b128(&lds_k[bi][kc_row * 32 + kc_c], gk);
    async_copy_b128(&lds_k[bi][kc_row * 32 + kc_c + 8], gk + 8);
    const bf16* gv =
        vT + ((size_t)(b * 1024 + h * 64 + vc_d) << 11) + kv0 + vc_c;
    async_copy_b128(&lds_v[bi][vc_d * 64 + vc_c], gv);
    async_copy_b128(&lds_v[bi][vc_d * 64 + vc_c + 8], gv + 8);
    async_copy_b128(&lds_v[bi][vc_d * 64 + vc_c + 16], gv + 16);
    async_copy_b128(&lds_v[bi][vc_d * 64 + vc_c + 24], gv + 24);
  };

  v8f zero = {};
  v8f o[4];
  float m_run[8], l_run[8];
#pragma unroll
  for (int jd = 0; jd < 4; ++jd) o[jd] = zero;
#pragma unroll
  for (int r = 0; r < 8; ++r) { m_run[r] = -1e30f; l_run[r] = 0.f; }

  issue(0, 0);
  wait_async();
  __syncthreads();

  const int nkv = SEQ / 64;
  for (int it = 0; it < nkv; ++it) {
    const int bi = it & 1;
    if (it + 1 < nkv) issue((it + 1) * 64, bi ^ 1);

    // logits S(16x64) = Qf @ K^T, scale 1/sqrt(64)
    v8f s[4];
#pragma unroll
    for (int j = 0; j < 4; ++j)
      s[j] = wmma_bf16(qf, frag_ld(&lds_k[bi][j * 16 * 32], 32, lane), zero);
#pragma unroll
    for (int j = 0; j < 4; ++j)
#pragma unroll
      for (int r = 0; r < 8; ++r) s[j][r] *= 0.125f;

    // row max across 64 keys
    float mx[8];
#pragma unroll
    for (int r = 0; r < 8; ++r)
      mx[r] = fmaxf(fmaxf(s[0][r], s[1][r]), fmaxf(s[2][r], s[3][r]));
#pragma unroll
    for (int off = 8; off >= 1; off >>= 1)
#pragma unroll
      for (int r = 0; r < 8; ++r)
        mx[r] = fmaxf(mx[r], __shfl_xor(mx[r], off, 32));

    // online softmax
    float l_new[8];
#pragma unroll
    for (int r = 0; r < 8; ++r) {
      const float mn = fmaxf(m_run[r], mx[r]);
      const float sc = __expf(m_run[r] - mn);
      m_run[r] = mn;
      l_run[r] *= sc;
#pragma unroll
      for (int jd = 0; jd < 4; ++jd) o[jd][r] *= sc;
      float ls = 0.f;
#pragma unroll
      for (int j = 0; j < 4; ++j) {
        const float p = __expf(s[j][r] - mn);
        s[j][r] = p;
        ls += p;
      }
      l_new[r] = ls;
    }
#pragma unroll
    for (int off = 8; off >= 1; off >>= 1)
#pragma unroll
      for (int r = 0; r < 8; ++r) l_new[r] += __shfl_xor(l_new[r], off, 32);
#pragma unroll
    for (int r = 0; r < 8; ++r) l_run[r] += l_new[r];

    // spill P (per-wave region; same-wave LDS ops are hw-ordered)
    bf16* pw = &lds_p[wave][0];
#pragma unroll
    for (int j = 0; j < 4; ++j)
#pragma unroll
      for (int r = 0; r < 8; ++r)
        pw[(mhalf + r) * 64 + j * 16 + ncol] = (bf16)s[j][r];
    lds_fence();

    // O += P(16x64) @ V(64x64)
    const v16bf a0 = frag_ld(pw, 64, lane);
    const v16bf a1 = frag_ld(pw + 32, 64, lane);
#pragma unroll
    for (int jd = 0; jd < 4; ++jd) {
      const bf16* vb = &lds_v[bi][jd * 16 * 64];
      o[jd] = wmma_bf16(a0, frag_ld(vb, 64, lane), o[jd]);
      o[jd] = wmma_bf16(a1, frag_ld(vb + 32, 64, lane), o[jd]);
    }

    wait_async();
    __syncthreads();
  }

#pragma unroll
  for (int jd = 0; jd < 4; ++jd)
#pragma unroll
    for (int r = 0; r < 8; ++r) {
      const size_t row = (size_t)(b * SEQ + q0 + wave * 16 + mhalf + r);
      out[row * 1024 + h * 64 + jd * 16 + ncol] = o[jd][r] / l_run[r];
    }
}

// ---------------------------------------------------------------------------
// LayerNorm: one block (256 thr) per row of 1024.
// ---------------------------------------------------------------------------
template <typename OT>
__global__ __launch_bounds__(256) void layernorm_k(
    const float* __restrict__ in, const float* __restrict__ gamma,
    const float* __restrict__ beta, OT* __restrict__ out) {
  __shared__ float s1[256], s2[256];
  const int row = blockIdx.x, t = threadIdx.x;
  const float* x = in + (size_t)row * DMODEL;
  float v[4], a = 0.f, b = 0.f;
#pragma unroll
  for (int i = 0; i < 4; ++i) {
    v[i] = x[t + i * 256];
    a += v[i];
    b += v[i] * v[i];
  }
  s1[t] = a;
  s2[t] = b;
  __syncthreads();
  for (int s = 128; s > 0; s >>= 1) {
    if (t < s) {
      s1[t] += s1[t + s];
      s2[t] += s2[t + s];
    }
    __syncthreads();
  }
  const float mu = s1[0] * (1.f / DMODEL);
  const float var = s2[0] * (1.f / DMODEL) - mu * mu;
  const float rs = rsqrtf(var + 1e-3f);
  OT* y = out + (size_t)row * DMODEL;
#pragma unroll
  for (int i = 0; i < 4; ++i) {
    const int c = t + i * 256;
    y[c] = (OT)((v[i] - mu) * rs * gamma[c] + beta[c]);
  }
}

// ---------------------------------------------------------------------------
extern "C" void kernel_launch(void* const* d_in, const int* in_sizes, int n_in,
                              void* d_out, int out_size, void* d_ws,
                              size_t ws_size, hipStream_t stream) {
  (void)in_sizes; (void)n_in; (void)out_size; (void)ws_size;
  const float* x   = (const float*)d_in[0];
  const float* Wq  = (const float*)d_in[1];
  const float* bq  = (const float*)d_in[2];
  const float* Wk  = (const float*)d_in[3];
  const float* bk  = (const float*)d_in[4];
  const float* Wv  = (const float*)d_in[5];
  const float* bv  = (const float*)d_in[6];
  const float* g1  = (const float*)d_in[7];
  const float* be1 = (const float*)d_in[8];
  const float* Wfc = (const float*)d_in[9];
  const float* bfc = (const float*)d_in[10];
  const float* g2  = (const float*)d_in[11];
  const float* be2 = (const float*)d_in[12];
  float* out = (float*)d_out;

  char* ws = (char*)d_ws;
  bf16* xb    = (bf16*)(ws);
  bf16* qb    = (bf16*)(ws + 8 * MB);
  bf16* kb    = (bf16*)(ws + 12 * MB);
  bf16* vT    = (bf16*)(ws + 16 * MB);
  bf16* WqT   = (bf16*)(ws + 24 * MB);
  bf16* WkT   = (bf16*)(ws + 25 * MB);
  bf16* WvT   = (bf16*)(ws + 26 * MB);
  bf16* WfcT  = (bf16*)(ws + 28 * MB);
  float* attn = (float*)(ws + 32 * MB);
  bf16* ln1   = (bf16*)(ws + 48 * MB);
  float* fc   = attn;

  const dim3 blk256(256);
  cvt_f32_bf16<<<dim3((MROWS * DMODEL) / 256), blk256, 0, stream>>>(
      x, xb, MROWS * DMODEL);
  transpose_f32_bf16<<<dim3((512 * 1024) / 256), blk256, 0, stream>>>(
      Wq, WqT, DMODEL, 512);
  transpose_f32_bf16<<<dim3((512 * 1024) / 256), blk256, 0, stream>>>(
      Wk, WkT, DMODEL, 512);
  transpose_f32_bf16<<<dim3((1024 * 1024) / 256), blk256, 0, stream>>>(
      Wv, WvT, DMODEL, 1024);
  transpose_f32_bf16<<<dim3((1024 * 1024) / 256), blk256, 0, stream>>>(
      Wfc, WfcT, DMODEL, 1024);

  gemm_tdm<bf16, false><<<dim3(4, 32), blk256, 0, stream>>>(
      xb, WqT, bq, qb, 512, DMODEL);
  gemm_tdm<bf16, false><<<dim3(4, 32), blk256, 0, stream>>>(
      xb, WkT, bk, kb, 512, DMODEL);
  gemm_tdm<bf16, true><<<dim3(8, 32), blk256, 0, stream>>>(
      xb, WvT, bv, vT, 1024, DMODEL);

  attn_wmma<<<dim3(SEQ / 64, NHEAD, BATCH), dim3(128), 0, stream>>>(qb, kb, vT,
                                                                    attn);
  layernorm_k<bf16><<<dim3(MROWS), blk256, 0, stream>>>(attn, g1, be1, ln1);
  gemm_tdm<float, false><<<dim3(8, 32), blk256, 0, stream>>>(
      ln1, WfcT, bfc, fc, 1024, DMODEL);
  layernorm_k<float><<<dim3(MROWS), blk256, 0, stream>>>(fc, g2, be2, out);
}
